// Model_1_36584531427773
// MI455X (gfx1250) — matrix operand order repaired and checked by reading, not compiled, not run on hardware
//
#include <hip/hip_runtime.h>
#include <cstdint>

// ---------------------------------------------------------------------------
// Types for CDNA5 WMMA (wave32): 16x16x32 f16 -> f32
// ---------------------------------------------------------------------------
typedef __attribute__((ext_vector_type(16))) _Float16 v16h;
typedef __attribute__((ext_vector_type(8)))  _Float16 v8h;
typedef __attribute__((ext_vector_type(8)))  float    v8f;
typedef __attribute__((ext_vector_type(4)))  unsigned int u32x4;
typedef __attribute__((ext_vector_type(4)))  int          i32x4;
typedef __attribute__((ext_vector_type(8)))  int          i32x8;

#define NB   64          // batch
#define NT   1024        // timesteps
#define DIN  212         // raw input features
#define TLn  35
#define PFn  57
#define NEX  6
#define KP   64          // padded feature dim (41 -> 64)
#define Hn   256         // hidden
#define G4   1024        // 4*H
#define ROWS (NB*NT)     // 65536

__device__ __forceinline__ float sigmoidf_(float x){ return 1.0f/(1.0f+__expf(-x)); }

// ---------------------------------------------------------------------------
// Fragment loaders, contiguous k order (not the ISA table order; A and B match).
// A (16x32 f16, MxK): lane<16 -> M=lane, K=0..15; lane>=16 -> K=16..31.
// ---------------------------------------------------------------------------
__device__ __forceinline__ v16h load_a_frag(const _Float16* A, int lda){
  const int lane = threadIdx.x & 31;
  const int row  = lane & 15;
  const int khi  = (lane >> 4) << 4;           // 0 or 16
  const _Float16* p = A + row*lda + khi;
  return *(const v16h*)(p);
}

// B (32x16 f16, KxN) loaded from a PRE-TRANSPOSED weight matrix Bt[N][Kld]:
// lane<16 -> N=lane, K=0..15; lane>=16 -> N=lane-16, K=16..31.  One 32B load.
__device__ __forceinline__ v16h load_bt_frag(const _Float16* Bt, int ldk){
  const int lane = threadIdx.x & 31;
  const int col  = lane & 15;
  const int kb   = (lane >> 4) << 4;           // 0 or 16
  return *(const v16h*)(Bt + (size_t)col*ldk + kb);
}

// ---------------------------------------------------------------------------
// TDM: stage one timestep of xz ([64 rows] x [1024 f16], row stride NT*1024
// elements) into LDS.  D# per CDNA5 ISA:
//   group0: count=1 | lds_addr | global_addr(57b) | type=2
//   group1: data_size=2B, tensor_dim0=1024, tensor_dim1=64,
//           tile_dim0=1024, tile_dim1=64, tensor_dim0_stride=NT*1024
// Issued by one wave; completion via s_wait_tensorcnt.
// ---------------------------------------------------------------------------
__device__ __forceinline__ void tdm_load_xz_step(const _Float16* xz, int t,
                                                 unsigned lds_off)
{
  unsigned long long ga = (unsigned long long)(uintptr_t)xz
                        + (unsigned long long)t * (G4*2);
  u32x4 g0;
  g0[0] = 1u;                                         // count=1 (valid user D#)
  g0[1] = lds_off;                                    // LDS byte address
  g0[2] = (unsigned)(ga & 0xffffffffu);               // global_addr[31:0]
  g0[3] = (unsigned)((ga >> 32) & 0x01ffffffu)        // global_addr[56:32]
        | (2u << 30);                                 // type=2 ("image")
  i32x8 g1;
  g1[0] = (int)(1u << 16);        // data_size=1 -> 2 bytes/elem
  g1[1] = (int)(1024u << 16);     // tensor_dim0[15:0]  @ bits[63:48]
  g1[2] = (int)(64u   << 16);     // tensor_dim1[15:0]  @ bits[95:80]
  g1[3] = (int)(1024u << 16);     // tile_dim0          @ bits[127:112]
  g1[4] = 64;                     // tile_dim1=64, tile_dim2=0
  g1[5] = (int)(NT * 1024);       // tensor_dim0_stride[31:0] (elem units)
  g1[6] = 0;
  g1[7] = 0;
  i32x4 z4 = {0,0,0,0};
#if __has_include(<hip/amd_detail/amd_gfx1250_TDM.h>)
  i32x8 z8 = {0,0,0,0,0,0,0,0};
  __builtin_amdgcn_tensor_load_to_lds(g0, g1, z4, z4, z8, 0);   // clang-23 arity
#else
  __builtin_amdgcn_tensor_load_to_lds(g0, g1, z4, z4, 0);       // ROCm 7.2 arity
#endif
}

// ---------------------------------------------------------------------------
// Weight prep: fp32 [K][N] row-major -> f16 [N][Kpad] (transposed, zero-pad K)
// ---------------------------------------------------------------------------
__global__ void transpose_cast(const float* __restrict__ W, _Float16* __restrict__ Wt,
                               int K, int N, int Kpad){
  int idx = blockIdx.x*blockDim.x + threadIdx.x;
  if (idx >= N*Kpad) return;
  int n = idx / Kpad, k = idx % Kpad;
  Wt[(size_t)n*Kpad + k] = (k < K) ? (_Float16)W[(size_t)k*N + n] : (_Float16)0.f;
}

// ---------------------------------------------------------------------------
// Tiny-MLP embedding front end: one (b,t) row per thread, weights in LDS.
// ---------------------------------------------------------------------------
__global__ void embed_kernel(const float* __restrict__ x,
    const float* __restrict__ pe_w1, const float* __restrict__ pe_b1,
    const float* __restrict__ pe_w2, const float* __restrict__ pe_b2,
    const float* __restrict__ te_w1, const float* __restrict__ te_b1,
    const float* __restrict__ te_w2, const float* __restrict__ te_b2,
    _Float16* __restrict__ feat)
{
  __shared__ float sw[TLn*17 + 17 + 17*5 + 5 + PFn*27 + 27 + 27*10 + 10];
  float* s_tw1 = sw;
  float* s_tb1 = s_tw1 + TLn*17;
  float* s_tw2 = s_tb1 + 17;
  float* s_tb2 = s_tw2 + 17*5;
  float* s_pw1 = s_tb2 + 5;
  float* s_pb1 = s_pw1 + PFn*27;
  float* s_pw2 = s_pb1 + 27;
  float* s_pb2 = s_pw2 + 27*10;
  for (int i=threadIdx.x; i<TLn*17; i+=blockDim.x) s_tw1[i]=te_w1[i];
  for (int i=threadIdx.x; i<17;     i+=blockDim.x) s_tb1[i]=te_b1[i];
  for (int i=threadIdx.x; i<17*5;   i+=blockDim.x) s_tw2[i]=te_w2[i];
  for (int i=threadIdx.x; i<5;      i+=blockDim.x) s_tb2[i]=te_b2[i];
  for (int i=threadIdx.x; i<PFn*27; i+=blockDim.x) s_pw1[i]=pe_w1[i];
  for (int i=threadIdx.x; i<27;     i+=blockDim.x) s_pb1[i]=pe_b1[i];
  for (int i=threadIdx.x; i<27*10;  i+=blockDim.x) s_pw2[i]=pe_w2[i];
  for (int i=threadIdx.x; i<10;     i+=blockDim.x) s_pb2[i]=pe_b2[i];
  __syncthreads();

  int r = blockIdx.x*blockDim.x + threadIdx.x;
  if (r >= ROWS) return;
  const float* xr = x + (size_t)r*DIN;
  _Float16* fr = feat + (size_t)r*KP;

  float th[17];
  for (int j=0;j<17;++j){
    float s = s_tb1[j];
    for (int k=0;k<TLn;++k) s += xr[k]*s_tw1[k*17+j];
    th[j] = fmaxf(s, 0.f);
  }
  for (int j=0;j<5;++j){
    float s = s_tb2[j];
    for (int k=0;k<17;++k) s += th[k]*s_tw2[k*5+j];
    fr[j] = (_Float16)fmaxf(s, 0.f);
  }
  for (int p=0;p<3;++p){
    const float* pp = xr + TLn + p*PFn;
    float ph[27];
    for (int j=0;j<27;++j){
      float s = s_pb1[j];
      for (int k=0;k<PFn;++k) s += pp[k]*s_pw1[k*27+j];
      ph[j] = fmaxf(s, 0.f);
    }
    for (int j=0;j<10;++j){
      float s = s_pb2[j];
      for (int k=0;k<27;++k) s += ph[k]*s_pw2[k*10+j];
      fr[5 + p*10 + j] = (_Float16)fmaxf(s, 0.f);
    }
  }
  for (int e=0;e<NEX;++e) fr[35+e] = (_Float16)xr[TLn + 3*PFn + e];
  for (int z=41; z<KP; ++z) fr[z] = (_Float16)0.f;
}

// ---------------------------------------------------------------------------
// WMMA GEMM: C[M][1024](f16) = A[M][K](f16) @ Bt^T + bias(f32).
// Block = 8 waves; wave w owns M-tile blockIdx.y*8+w (16 rows) x 64 columns.
// ---------------------------------------------------------------------------
__global__ __launch_bounds__(256)
void wmma_gemm_bias(const _Float16* __restrict__ A, const _Float16* __restrict__ Bt,
                    const float* __restrict__ bias, _Float16* __restrict__ C, int K)
{
  const int wave  = threadIdx.x >> 5;
  const int lane  = threadIdx.x & 31;
  const int mtile = blockIdx.y*8 + wave;
  const int nbase = blockIdx.x*64;
  const _Float16* Arow = A + (size_t)mtile*16*K;

  v8f acc[4];
  {
    const float b0 = bias[nbase +  0 + (lane&15)];
    const float b1 = bias[nbase + 16 + (lane&15)];
    const float b2 = bias[nbase + 32 + (lane&15)];
    const float b3 = bias[nbase + 48 + (lane&15)];
    #pragma unroll
    for (int r=0;r<8;++r){ acc[0][r]=b0; acc[1][r]=b1; acc[2][r]=b2; acc[3][r]=b3; }
  }
  for (int kk=0; kk<K; kk+=32){
    v16h a = load_a_frag(Arow + kk, K);
    #pragma unroll
    for (int i=0;i<4;++i){
      v16h b = load_bt_frag(Bt + (size_t)(nbase + i*16)*K + kk, K);
      acc[i] = __builtin_amdgcn_wmma_f32_16x16x32_f16(false, a, false, b,
                                                      (short)0, acc[i], false, false);
    }
  }
  const int rb   = (lane >> 4) << 3;
  const int col0 = lane & 15;
  #pragma unroll
  for (int i=0;i<4;++i){
    _Float16* Cp = C + (size_t)mtile*16*G4 + (nbase + i*16 + col0);
    #pragma unroll
    for (int r=0;r<8;++r)
      Cp[(size_t)(rb + r)*G4] = (_Float16)acc[i][r];
  }
}

// ---------------------------------------------------------------------------
// Row softmax statistics over a [64][256] LDS buffer.
// ---------------------------------------------------------------------------
__device__ __forceinline__ void rowstats_softmax(const float* buf, float* pred,
                                                 float* rowm, float* rsum, int tid){
  const int row = tid >> 4, seg = tid & 15;
  const float* gr = buf + row*Hn + seg*16;
  float m = gr[0];
  #pragma unroll
  for (int k=1;k<16;++k) m = fmaxf(m, gr[k]);
  pred[tid] = m;
  __syncthreads();
  if (tid < 64){
    float mm = pred[tid*16];
    for (int k=1;k<16;++k) mm = fmaxf(mm, pred[tid*16+k]);
    rowm[tid] = mm;
  }
  __syncthreads();
  const float mm = rowm[row];
  float s = 0.f;
  #pragma unroll
  for (int k=0;k<16;++k) s += __expf(gr[k]-mm);
  pred[tid] = s;
  __syncthreads();
  if (tid < 64){
    float ss = 0.f;
    for (int k=0;k<16;++k) ss += pred[tid*16+k];
    rsum[tid] = ss;
  }
  __syncthreads();
}

// ---------------------------------------------------------------------------
// Persistent single-WGP LSTM scan.  32 wave32 waves; each wave owns two
// (mtile,ntile) pairs of the 4x16 tile grid.  xz[t] is DMA'd into LDS by the
// Tensor Data Mover one step ahead (overlapped with the softmax phase); h is
// LDS-resident f16; U streams from the 192MB L2.  LDS: 128K stage + 164K
// state = 293KB of the WGP's 320KB.
// ---------------------------------------------------------------------------
__global__ __launch_bounds__(1024)
void lstm_scan_kernel(const _Float16* __restrict__ xz,    // [ROWS][1024] f16, row=b*NT+t
                      const _Float16* __restrict__ Ut,    // [1024][256] transposed
                      _Float16* __restrict__ hseq)        // [ROWS][256]
{
  extern __shared__ float smem[];
  float*    gbuf = smem;                        // 64*256 f32
  float*    cbuf = gbuf + NB*Hn;                // 64*256 f32
  float*    pred = cbuf + NB*Hn;                // 1024
  float*    rowm = pred + 1024;                 // 64
  float*    rsum = rowm + 64;                   // 64
  _Float16* hbuf = (_Float16*)(rsum + 64);      // 64*256 f16
  _Float16* xzs  = hbuf + NB*Hn;                // 64*1024 f16 TDM stage (128KB)

  const int tid  = threadIdx.x;
  const int wave = tid >> 5;
  const int lane = tid & 31;
  const int col0 = lane & 15;
  const int rb   = (lane >> 4) << 3;
  const unsigned xzs_lds = (unsigned)(uintptr_t)(void*)xzs;  // LDS byte offset

  if (wave == 0) tdm_load_xz_step(xz, 0, xzs_lds);   // prefetch t=0
  for (int i=tid; i<NB*Hn; i+=1024){ cbuf[i]=0.f; hbuf[i]=(_Float16)0.f; }
  __syncthreads();

  for (int t=0; t<NT; ++t){
    if (wave == 0) __builtin_amdgcn_s_wait_tensorcnt((short)0);
    __syncthreads();                                  // stage[t] visible to all

    float ik[2][8], fk[2][8], ok[2][8];
    #pragma unroll
    for (int pp=0; pp<2; ++pp){
      const int pair  = wave*2 + pp;
      const int mtile = pair >> 4;            // 0..3
      const int ntile = pair & 15;            // 0..15
      const int nb16  = ntile*16 + col0;

      v8f ai, af, ag, ao;
      #pragma unroll
      for (int r=0;r<8;++r){
        const int b = mtile*16 + rb + r;
        const _Float16* zr = xzs + (size_t)b*G4;      // staged in LDS
        ai[r] = (float)zr[      nb16];
        af[r] = (float)zr[256 + nb16];
        ag[r] = (float)zr[512 + nb16];
        ao[r] = (float)zr[768 + nb16];
      }
      const _Float16* Ah = hbuf + mtile*16*Hn;
      const _Float16* Ui = Ut + (size_t)(      ntile*16)*Hn;
      const _Float16* Uf = Ut + (size_t)(256 + ntile*16)*Hn;
      const _Float16* Ug = Ut + (size_t)(512 + ntile*16)*Hn;
      const _Float16* Uo = Ut + (size_t)(768 + ntile*16)*Hn;
      #pragma unroll
      for (int kk=0; kk<Hn; kk+=32){
        __builtin_prefetch(Ui + kk + 32, 0, 1);       // global_prefetch_b8
        v16h a  = load_a_frag(Ah + kk, Hn);           // h from LDS
        v16h bi = load_bt_frag(Ui + kk, Hn);
        v16h bf = load_bt_frag(Uf + kk, Hn);
        v16h bg = load_bt_frag(Ug + kk, Hn);
        v16h bo = load_bt_frag(Uo + kk, Hn);
        ai = __builtin_amdgcn_wmma_f32_16x16x32_f16(false,a,false,bi,(short)0,ai,false,false);
        af = __builtin_amdgcn_wmma_f32_16x16x32_f16(false,a,false,bf,(short)0,af,false,false);
        ag = __builtin_amdgcn_wmma_f32_16x16x32_f16(false,a,false,bg,(short)0,ag,false,false);
        ao = __builtin_amdgcn_wmma_f32_16x16x32_f16(false,a,false,bo,(short)0,ao,false,false);
      }
      #pragma unroll
      for (int r=0;r<8;++r){
        const int row = mtile*16 + rb + r;
        gbuf[row*Hn + nb16] = ag[r];
        ik[pp][r]=ai[r]; fk[pp][r]=af[r]; ok[pp][r]=ao[r];
      }
    }
    __syncthreads();                                  // all stage reads done

    if (wave == 0 && t+1 < NT)
      tdm_load_xz_step(xz, t+1, xzs_lds);             // DMA next step, overlapped

    rowstats_softmax(gbuf, pred, rowm, rsum, tid);    // softmax(g) stats

    #pragma unroll
    for (int pp=0; pp<2; ++pp){
      const int pair = wave*2 + pp;
      const int mtile = pair >> 4, ntile = pair & 15;
      const int nb16 = ntile*16 + col0;
      #pragma unroll
      for (int r=0;r<8;++r){
        const int row = mtile*16 + rb + r;
        const int idx = row*Hn + nb16;
        const float sg = __expf(gbuf[idx] - rowm[row]) / rsum[row];
        cbuf[idx] = sigmoidf_(fk[pp][r])*cbuf[idx] + sigmoidf_(ik[pp][r])*sg;
      }
    }
    __syncthreads();

    rowstats_softmax(cbuf, pred, rowm, rsum, tid);    // softmax(c) stats

    #pragma unroll
    for (int pp=0; pp<2; ++pp){
      const int pair = wave*2 + pp;
      const int mtile = pair >> 4, ntile = pair & 15;
      const int nb16 = ntile*16 + col0;
      #pragma unroll
      for (int r=0;r<8;++r){
        const int row = mtile*16 + rb + r;
        const int idx = row*Hn + nb16;
        const float sc = __expf(cbuf[idx] - rowm[row]) / rsum[row];
        const float hv = sigmoidf_(ok[pp][r]) * sc;
        hbuf[idx] = (_Float16)hv;
        hseq[((size_t)row*NT + t)*Hn + nb16] = (_Float16)hv;
      }
    }
    __syncthreads();
  }
}

// ---------------------------------------------------------------------------
// Dense head: one batch row per thread (64 threads), weights L2-resident.
// ---------------------------------------------------------------------------
__global__ void head_kernel(const _Float16* __restrict__ hseq,
  const float* __restrict__ d1w, const float* __restrict__ d1b,
  const float* __restrict__ d2w, const float* __restrict__ d2b,
  const float* __restrict__ d3w, const float* __restrict__ d3b,
  float* __restrict__ out)
{
  int b = threadIdx.x;
  if (b >= NB) return;
  const _Float16* h = hseq + ((size_t)b*NT + (NT-1))*Hn;
  float a1[100];
  for (int j=0;j<100;++j){
    float s = d1b[j];
    for (int k=0;k<Hn;++k) s += (float)h[k]*d1w[k*100+j];
    a1[j] = fmaxf(s, 0.f);
  }
  float a2[40];
  for (int j=0;j<40;++j){
    float s = d2b[j];
    for (int k=0;k<100;++k) s += a1[k]*d2w[k*40+j];
    a2[j] = fmaxf(s, 0.f);
  }
  float l0 = d3b[0], l1 = d3b[1];
  for (int k=0;k<40;++k){ l0 += a2[k]*d3w[k*2+0]; l1 += a2[k]*d3w[k*2+1]; }
  float m = fmaxf(l0, l1);
  float e0 = __expf(l0-m), e1 = __expf(l1-m);
  out[b*2+0] = e0/(e0+e1);
  out[b*2+1] = e1/(e0+e1);
}

// ---------------------------------------------------------------------------
extern "C" void kernel_launch(void* const* d_in, const int* in_sizes, int n_in,
                              void* d_out, int out_size, void* d_ws, size_t ws_size,
                              hipStream_t stream)
{
  (void)in_sizes; (void)n_in; (void)out_size; (void)ws_size;
  const float* x     = (const float*)d_in[0];
  const float* pe_w1 = (const float*)d_in[1];
  const float* pe_b1 = (const float*)d_in[2];
  const float* pe_w2 = (const float*)d_in[3];
  const float* pe_b2 = (const float*)d_in[4];
  const float* te_w1 = (const float*)d_in[5];
  const float* te_b1 = (const float*)d_in[6];
  const float* te_w2 = (const float*)d_in[7];
  const float* te_b2 = (const float*)d_in[8];
  const float* W1    = (const float*)d_in[9];
  const float* U1    = (const float*)d_in[10];
  const float* bl1   = (const float*)d_in[11];
  const float* W2    = (const float*)d_in[12];
  const float* U2    = (const float*)d_in[13];
  const float* bl2   = (const float*)d_in[14];
  const float* W3    = (const float*)d_in[15];
  const float* U3    = (const float*)d_in[16];
  const float* bl3   = (const float*)d_in[17];
  const float* d1w   = (const float*)d_in[18];
  const float* d1b   = (const float*)d_in[19];
  const float* d2w   = (const float*)d_in[20];
  const float* d2b   = (const float*)d_in[21];
  const float* d3w   = (const float*)d_in[22];
  const float* d3b   = (const float*)d_in[23];

  char* ws = (char*)d_ws;
  size_t off = 0;
  auto carve = [&](size_t bytes)->char* {
    char* p = ws + off; off += (bytes + 255) & ~(size_t)255; return p;
  };
  _Float16* feat = (_Float16*)carve((size_t)ROWS*KP*2);
  _Float16* W1t  = (_Float16*)carve((size_t)G4*KP*2);
  _Float16* W2t  = (_Float16*)carve((size_t)G4*Hn*2);
  _Float16* W3t  = (_Float16*)carve((size_t)G4*Hn*2);
  _Float16* U1t  = (_Float16*)carve((size_t)G4*Hn*2);
  _Float16* U2t  = (_Float16*)carve((size_t)G4*Hn*2);
  _Float16* U3t  = (_Float16*)carve((size_t)G4*Hn*2);
  _Float16* xz   = (_Float16*)carve((size_t)ROWS*G4*2);
  _Float16* hA   = (_Float16*)carve((size_t)ROWS*Hn*2);
  _Float16* hB   = (_Float16*)carve((size_t)ROWS*Hn*2);

  // 1) weight prep (f16, transposed, K-padded)
  transpose_cast<<<dim3((G4*KP + 255)/256), 256, 0, stream>>>(W1, W1t, 41, G4, KP);
  transpose_cast<<<dim3((G4*Hn + 255)/256), 256, 0, stream>>>(W2, W2t, Hn, G4, Hn);
  transpose_cast<<<dim3((G4*Hn + 255)/256), 256, 0, stream>>>(W3, W3t, Hn, G4, Hn);
  transpose_cast<<<dim3((G4*Hn + 255)/256), 256, 0, stream>>>(U1, U1t, Hn, G4, Hn);
  transpose_cast<<<dim3((G4*Hn + 255)/256), 256, 0, stream>>>(U2, U2t, Hn, G4, Hn);
  transpose_cast<<<dim3((G4*Hn + 255)/256), 256, 0, stream>>>(U3, U3t, Hn, G4, Hn);

  // 2) embedding front end
  embed_kernel<<<dim3(ROWS/256), 256, 0, stream>>>(x, pe_w1, pe_b1, pe_w2, pe_b2,
                                                   te_w1, te_b1, te_w2, te_b2, feat);

  const dim3 ggrid(G4/64, ROWS/128);
  const size_t smem = (size_t)(NB*Hn*2 + 1024 + 64 + 64)*4     // g/c/red f32
                    + (size_t)NB*Hn*2                           // h f16
                    + (size_t)NB*G4*2;                          // xz TDM stage
  // = 299,520 B of the WGP's 320KB LDS

  // 3) layer 1: xz = feat@W1+b, then scan
  wmma_gemm_bias<<<ggrid, 256, 0, stream>>>(feat, W1t, bl1, xz, KP);
  lstm_scan_kernel<<<1, 1024, smem, stream>>>(xz, U1t, hA);
  // 4) layer 2
  wmma_gemm_bias<<<ggrid, 256, 0, stream>>>(hA, W2t, bl2, xz, Hn);
  lstm_scan_kernel<<<1, 1024, smem, stream>>>(xz, U2t, hB);
  // 5) layer 3
  wmma_gemm_bias<<<ggrid, 256, 0, stream>>>(hB, W3t, bl3, xz, Hn);
  lstm_scan_kernel<<<1, 1024, smem, stream>>>(xz, U3t, hA);
  // 6) dense head + softmax
  head_kernel<<<1, 64, 0, stream>>>(hA, d1w, d1b, d2w, d2b, d3w, d3b, (float*)d_out);
}
